// CompatibleAttentionWeatherGRU_36979668418760
// MI455X (gfx1250) — matrix operand order repaired and checked by reading, not compiled, not run on hardware
//
#include <hip/hip_runtime.h>
#include <hip/hip_bf16.h>
#include <math.h>

typedef __attribute__((ext_vector_type(16))) __bf16 v16bf;
typedef __attribute__((ext_vector_type(8)))  __bf16 v8bf;
typedef __attribute__((ext_vector_type(8)))  float  v8f;
typedef __attribute__((ext_vector_type(4)))  unsigned int v4u;
typedef __attribute__((ext_vector_type(8)))  int v8i;
typedef __attribute__((ext_vector_type(4)))  int v4i;

#define HD 1024
#define BB 32
#define SS 128
#define TT 64
#define VV 32000
#define KC 128               // K elements staged per TDM chunk
#define LDS_ROW (KC + 8)     // TDM pads 4 DWORDs (8 bf16) per 64-DWORD row -> 272B stride

// ---------------- utility kernels ----------------

__global__ void k_f32_to_bf16(const float* __restrict__ src, __bf16* __restrict__ dst, long n) {
    long i = (long)blockIdx.x * blockDim.x + threadIdx.x;
    long stride = (long)gridDim.x * blockDim.x;
    for (; i < n; i += stride) dst[i] = (__bf16)src[i];
}

__global__ void k_zero_h(float* __restrict__ h, __bf16* __restrict__ hb, int n) {
    int i = blockIdx.x * blockDim.x + threadIdx.x;
    if (i < n) { h[i] = 0.f; hb[i] = (__bf16)0.f; }
}

// outputs[:,0,:] = one-hot(target_tokens[0,0])
__global__ void k_init_out(float* __restrict__ out, const int* __restrict__ tok) {
    long i = (long)blockIdx.x * blockDim.x + threadIdx.x;
    if (i >= (long)BB * VV) return;
    int b = (int)(i / VV), v = (int)(i % VV);
    out[(size_t)b * TT * VV + v] = (v == tok[0]) ? 1.f : 0.f;
}

// ---------------- feature concat (tiny projections fused, emit bf16) ----------------

__global__ void k_build_concat(const float* __restrict__ feat,
                               const float* __restrict__ tW, const float* __restrict__ tb,
                               const float* __restrict__ hW, const float* __restrict__ hb_,
                               const float* __restrict__ cW, const float* __restrict__ cb,
                               const float* __restrict__ timW, const float* __restrict__ timb,
                               const float* __restrict__ oW, const float* __restrict__ ob,
                               __bf16* __restrict__ out) {
    long i = (long)blockIdx.x * blockDim.x + threadIdx.x;
    long n = (long)BB * SS * 2048;
    if (i >= n) return;
    int c = (int)(i & 2047);
    long row = i >> 11;
    const float* f = feat + row * 16;
    float v;
    if (c < 256)       { v = f[0] * tW[c] + tb[c]; }
    else if (c < 512)  { int q = c - 256; v = f[1] * hW[q] + hb_[q]; }
    else if (c < 768)  { int q = c - 512; v = f[2] * cW[q] + cb[q]; }
    else if (c < 1024) { int q = c - 768; v = f[11] * timW[2*q] + f[12] * timW[2*q+1] + timb[q]; }
    else {
        int q = c - 1024; const float* w = oW + q * 11;
        v = ob[q];
        #pragma unroll
        for (int j = 0; j < 8; ++j) v += f[3 + j] * w[j];
        v += f[13] * w[8] + f[14] * w[9] + f[15] * w[10];
    }
    out[i] = (__bf16)v;
}

// ---------------- TDM: stage a 64-row x KC-col bf16 tile of W into LDS ----------------
// D# (ISA 8.3/8.4): group0 = {count|flags, lds_addr, global_addr_lo, global_addr_hi|type},
// group1 packs data_size=2B, pad(4 DW per 64 DW row), tensor/tile dims + dim0 stride.
// This toolchain exposes the 6-arg builtin: (v4u, v8i, v4i, v4i, v8i, cpol).

__device__ inline void tdm_load_tile(const __bf16* gptr, unsigned lds_off, int K, int N) {
    unsigned long long ga = (unsigned long long)gptr;
    v4u g0;
    g0.x = 1u;                                             // count=1, user descriptor
    g0.y = lds_off;                                        // LDS byte address
    g0.z = (unsigned)ga;                                   // global_addr[31:0]
    g0.w = (unsigned)((ga >> 32) & 0x01FFFFFFull) | (2u << 30);  // addr[56:32] | type=2
    v8i g1;
    g1[0] = (int)((1u << 16)      // data_size = 1 -> 2 bytes
                | (1u << 20)      // pad_enable
                | (5u << 22)      // pad_interval code 5 -> every 64 DWORDs (one row)
                | (3u << 25));    // pad_amount code 3 -> 4 DWORDs (16B) -> 272B row stride
    g1[1] = (int)(((unsigned)K & 0xFFFFu) << 16);                    // tensor_dim0[15:0]
    g1[2] = (int)(((unsigned)K >> 16) | (((unsigned)N & 0xFFFFu) << 16)); // td0 hi | td1 lo
    g1[3] = (int)(((unsigned)N >> 16) | ((unsigned)KC << 16));       // td1 hi | tile_dim0=128
    g1[4] = 64;                                                      // tile_dim1=64, tile_dim2=0
    g1[5] = K;                                                       // tensor_dim0_stride lo
    g1[6] = 0;                                                       // stride hi | td1_stride lo
    g1[7] = 0;
    v4i z4 = {0, 0, 0, 0};
    v8i z8 = {0, 0, 0, 0, 0, 0, 0, 0};
    __builtin_amdgcn_tensor_load_to_lds(g0, g1, z4, z4, z8, 0);
}

// ---------------- WMMA bf16 GEMM: C[M,N] = act(A[M,K] @ W[N,K]^T + bias) ----------------
// 8 waves/block, wave -> 16x64 strip. W tiles double-buffered in LDS via TDM (wave 0 issues,
// s_wait_tensorcnt + barriers). A fragments are direct 16B global loads (small, L2-resident).
// All 4 B fragments are prefetched into distinct registers so the 4 WMMAs issue back-to-back.
// scatterB>0: row m=t*B+b of C is written to outF[b, t+1, :] of a [B,T,N] tensor.

__global__ void k_gemm_bf16(const __bf16* __restrict__ A, const __bf16* __restrict__ W,
                            const float* __restrict__ bias,
                            float* __restrict__ outF, __bf16* __restrict__ outB,
                            int M, int N, int K, int act, int scatterB, int scatterT) {
    __shared__ __bf16 sW[2][64 * LDS_ROW];

    int lane = threadIdx.x & 31;
    int wv   = threadIdx.x >> 5;
    int mt   = blockIdx.y * 8 + wv;
    bool active = (mt * 16 < M);
    int n0  = blockIdx.x * 64;
    int row = lane & 15, half = lane >> 4;

    const __bf16* Arow = A + (size_t)((active ? mt : 0) * 16 + row) * K;
    const __bf16* Wtile = W + (size_t)n0 * K;
    v8f acc[4] = {};

    int nkc = K / KC;
    if (wv == 0) tdm_load_tile(Wtile, (unsigned)(unsigned long long)&sW[0][0], K, N);

    for (int kc = 0; kc < nkc; ++kc) {
        if (wv == 0) {
            if (kc + 1 < nkc) {
                tdm_load_tile(Wtile + (size_t)(kc + 1) * KC,
                              (unsigned)(unsigned long long)&sW[(kc + 1) & 1][0], K, N);
                __builtin_amdgcn_s_wait_tensorcnt((short)1);   // chunk kc has landed
            } else {
                __builtin_amdgcn_s_wait_tensorcnt((short)0);
            }
        }
        __syncthreads();                                        // tile kc visible to all waves
        if (active) {
            const __bf16* Ws = &sW[kc & 1][0];
            #pragma unroll
            for (int kk = 0; kk < KC; kk += 32) {
                // A fragment (16x32): halves hold K=8h+0..7 / 16+8h+0..7
                v8bf lo = *(const v8bf*)(Arow + kc * KC + kk + 8 * half);
                v8bf hi = *(const v8bf*)(Arow + kc * KC + kk + 16 + 8 * half);
                v16bf a;
                #pragma unroll
                for (int i = 0; i < 8; ++i) { a[i] = lo[i]; a[8 + i] = hi[i]; }
                // Prefetch all 4 B fragments (padded rows -> staggered banks), then 4 WMMAs
                v16bf bfr[4];
                #pragma unroll
                for (int t = 0; t < 4; ++t) {
                    const __bf16* Wr = Ws + (16 * t + row) * LDS_ROW + kk + 8 * half;
                    v8bf b0 = *(const v8bf*)Wr;
                    v8bf b1 = *(const v8bf*)(Wr + 16);
                    #pragma unroll
                    for (int i = 0; i < 8; ++i) { bfr[t][i] = b0[i]; bfr[t][8 + i] = b1[i]; }
                }
                #pragma unroll
                for (int t = 0; t < 4; ++t) {
                    acc[t] = __builtin_amdgcn_wmma_f32_16x16x32_bf16(
                        false, a, false, bfr[t], (short)0, acc[t], false, false);
                }
            }
        }
        __syncthreads();                                        // done reading before overwrite
    }

    if (!active) return;
    #pragma unroll
    for (int t = 0; t < 4; ++t) {
        int n = n0 + 16 * t + row;               // C/D: lane&15 = N column
        float bv = bias ? bias[n] : 0.f;
        #pragma unroll
        for (int r = 0; r < 8; ++r) {
            int m = mt * 16 + r + 8 * half;      // C/D: VGPR r -> M = r + 8*(lane>=16)
            float v = acc[t][r] + bv;
            if (act) v = tanhf(v);
            size_t off;
            if (scatterB > 0) {
                int b_ = m % scatterB, tr = m / scatterB;
                off = ((size_t)b_ * scatterT + tr + 1) * (size_t)N + n;
            } else {
                off = (size_t)m * N + n;
            }
            if (outF) outF[off] = v;
            if (outB) outB[off] = (__bf16)v;
        }
    }
}

// ---------------- GRU gate combine (shared by encoder/decoder) ----------------

__global__ void k_gru_combine(const float* __restrict__ gi, int giStride,
                              const float* __restrict__ gh,
                              float* __restrict__ h, __bf16* __restrict__ hb,
                              float* __restrict__ o32, __bf16* __restrict__ ob16, int ors) {
    int i = blockIdx.x * blockDim.x + threadIdx.x;
    if (i >= BB * HD) return;
    int b = i >> 10, j = i & 1023;
    const float* gir = gi + (size_t)b * giStride;
    const float* ghr = gh + (size_t)b * 3 * HD;
    float r = 1.f / (1.f + expf(-(gir[j]          + ghr[j])));
    float z = 1.f / (1.f + expf(-(gir[HD + j]     + ghr[HD + j])));
    float n = tanhf(gir[2 * HD + j] + r * ghr[2 * HD + j]);
    float hn = (1.f - z) * n + z * h[i];
    h[i] = hn; hb[i] = (__bf16)hn;
    if (o32)  o32[(size_t)b * ors + j]  = hn;
    if (ob16) ob16[(size_t)b * ors + j] = (__bf16)hn;
}

// ---------------- attention: scores -> softmax -> context (one block per batch row) ----------------

__global__ void k_attn(const float* __restrict__ h, const float* __restrict__ keys,
                       const float* __restrict__ enc_out, float* __restrict__ ctx) {
    __shared__ float w[SS];
    int b = blockIdx.x;
    int lane = threadIdx.x & 31, wv = threadIdx.x >> 5;  // 4 waves
    const float* hb = h + (size_t)b * HD;
    for (int ts = wv; ts < SS; ts += 4) {
        const float* kr = keys + ((size_t)b * SS + ts) * HD;
        float s = 0.f;
        for (int j = lane; j < HD; j += 32) s += hb[j] * kr[j];
        #pragma unroll
        for (int o = 16; o >= 1; o >>= 1) s += __shfl_xor(s, o, 32);
        if (lane == 0) w[ts] = s * 0.03125f;  // 1/sqrt(1024)
    }
    __syncthreads();
    if (threadIdx.x == 0) {
        float mx = w[0];
        for (int s = 1; s < SS; ++s) mx = fmaxf(mx, w[s]);
        float sum = 0.f;
        for (int s = 0; s < SS; ++s) { w[s] = expf(w[s] - mx); sum += w[s]; }
        float inv = 1.f / sum;
        for (int s = 0; s < SS; ++s) w[s] *= inv;
    }
    __syncthreads();
    for (int f = threadIdx.x; f < HD; f += blockDim.x) {
        float acc = 0.f;
        const float* er = enc_out + (size_t)b * SS * HD + f;
        for (int s = 0; s < SS; ++s) acc += w[s] * er[(size_t)s * HD];
        ctx[(size_t)b * HD + f] = acc;
    }
}

// ---------------- decoder input build: x = [emb[tok] | ctx] (bf16); stash ctx into hc ----------------

__global__ void k_build_x(const int* __restrict__ tok, int step,
                          const float* __restrict__ emb, const float* __restrict__ ctx,
                          __bf16* __restrict__ x, __bf16* __restrict__ hc) {
    int i = blockIdx.x * blockDim.x + threadIdx.x;
    if (i >= BB * 2048) return;
    int b = i >> 11, j = i & 2047;
    float v;
    if (j < HD) {
        int t = tok[b * TT + step];
        v = emb[(size_t)t * HD + j];
    } else {
        v = ctx[(size_t)b * HD + (j - HD)];
        hc[(size_t)b * 2048 + j] = (__bf16)v;
    }
    x[(size_t)b * 2048 + j] = (__bf16)v;
}

// ---------------- host orchestration ----------------

extern "C" void kernel_launch(void* const* d_in, const int* in_sizes, int n_in,
                              void* d_out, int out_size, void* d_ws, size_t ws_size,
                              hipStream_t stream) {
    const float* features = (const float*)d_in[0];
    const int*   tokens   = (const int*)d_in[1];
    const float* temp_W = (const float*)d_in[2],  *temp_b = (const float*)d_in[3];
    const float* hum_W  = (const float*)d_in[4],  *hum_b  = (const float*)d_in[5];
    const float* cloud_W= (const float*)d_in[6],  *cloud_b= (const float*)d_in[7];
    const float* time_W = (const float*)d_in[8],  *time_b = (const float*)d_in[9];
    const float* other_W= (const float*)d_in[10], *other_b= (const float*)d_in[11];
    const float* fi_W   = (const float*)d_in[12], *fi_b   = (const float*)d_in[13];
    const float* enc_wih= (const float*)d_in[14], *enc_whh= (const float*)d_in[15];
    const float* enc_bih= (const float*)d_in[16], *enc_bhh= (const float*)d_in[17];
    const float* emb    = (const float*)d_in[18];
    const float* attn_W = (const float*)d_in[19], *attn_b = (const float*)d_in[20];
    const float* dec_wih= (const float*)d_in[21], *dec_whh= (const float*)d_in[22];
    const float* dec_bih= (const float*)d_in[23], *dec_bhh= (const float*)d_in[24];
    const float* out_W  = (const float*)d_in[25], *out_b  = (const float*)d_in[26];
    float* out = (float*)d_out;

    char* ws = (char*)d_ws;
    size_t off = 0;
    auto alloc = [&](size_t bytes) -> void* {
        void* p = ws + off; off += (bytes + 255) & ~(size_t)255; return p;
    };

    __bf16* fiW_b    = (__bf16*)alloc((size_t)HD * 2048 * 2);
    __bf16* encWih_b = (__bf16*)alloc((size_t)3 * HD * HD * 2);
    __bf16* encWhh_b = (__bf16*)alloc((size_t)3 * HD * HD * 2);
    __bf16* attnW_b  = (__bf16*)alloc((size_t)HD * HD * 2);
    __bf16* decWih_b = (__bf16*)alloc((size_t)3 * HD * 2 * HD * 2);
    __bf16* decWhh_b = (__bf16*)alloc((size_t)3 * HD * HD * 2);
    __bf16* outW_b   = (__bf16*)alloc((size_t)VV * 2 * HD * 2);
    __bf16* concat_b = (__bf16*)alloc((size_t)BB * SS * 2048 * 2);
    __bf16* proj_b   = (__bf16*)alloc((size_t)BB * SS * HD * 2);
    float*  gi_enc   = (float*) alloc((size_t)BB * SS * 3 * HD * 4);
    float*  enc_out  = (float*) alloc((size_t)BB * SS * HD * 4);
    __bf16* enc_out_b= (__bf16*)alloc((size_t)BB * SS * HD * 2);
    float*  keys     = (float*) alloc((size_t)BB * SS * HD * 4);
    float*  hst      = (float*) alloc((size_t)BB * HD * 4);
    __bf16* h_b      = (__bf16*)alloc((size_t)BB * HD * 2);
    float*  gh       = (float*) alloc((size_t)BB * 3 * HD * 4);
    float*  ctx      = (float*) alloc((size_t)BB * HD * 4);
    __bf16* x_b      = (__bf16*)alloc((size_t)BB * 2048 * 2);
    float*  gi_d     = (float*) alloc((size_t)BB * 3 * HD * 4);
    float*  gh_d     = (float*) alloc((size_t)BB * 3 * HD * 4);
    __bf16* hc_b     = (__bf16*)alloc((size_t)(TT - 1) * BB * 2048 * 2);

    auto cvt = [&](const float* s, __bf16* d, long n) {
        long blocks = (n + 255) / 256; if (blocks > 65535) blocks = 65535;
        k_f32_to_bf16<<<(int)blocks, 256, 0, stream>>>(s, d, n);
    };
    auto gemm = [&](const __bf16* A, const __bf16* Wm, const float* bias,
                    float* oF, __bf16* oB, int M, int N, int K, int act,
                    int sB, int sT) {
        dim3 g(N / 64, (M / 16 + 7) / 8);
        k_gemm_bf16<<<g, 256, 0, stream>>>(A, Wm, bias, oF, oB, M, N, K, act, sB, sT);
    };

    // 0) weight conversions (bf16 WMMA operands)
    cvt(fi_W,    fiW_b,    (long)HD * 2048);
    cvt(enc_wih, encWih_b, (long)3 * HD * HD);
    cvt(enc_whh, encWhh_b, (long)3 * HD * HD);
    cvt(attn_W,  attnW_b,  (long)HD * HD);
    cvt(dec_wih, decWih_b, (long)3 * HD * 2 * HD);
    cvt(dec_whh, decWhh_b, (long)3 * HD * HD);
    cvt(out_W,   outW_b,   (long)VV * 2 * HD);

    // 1) feature projections -> concat (bf16), then proj = tanh(concat @ fi_W^T + fi_b)
    k_build_concat<<<(BB * SS * 2048 + 255) / 256, 256, 0, stream>>>(
        features, temp_W, temp_b, hum_W, hum_b, cloud_W, cloud_b,
        time_W, time_b, other_W, other_b, concat_b);
    gemm(concat_b, fiW_b, fi_b, nullptr, proj_b, BB * SS, HD, 2048, 1, 0, 0);

    // 2) batched encoder input gates: gi = proj @ enc_wih^T + enc_bih   [4096, 3072]
    gemm(proj_b, encWih_b, enc_bih, gi_enc, nullptr, BB * SS, 3 * HD, HD, 0, 0, 0);

    // 3) encoder recurrence (whh stays L2-resident across 128 steps)
    k_zero_h<<<(BB * HD + 255) / 256, 256, 0, stream>>>(hst, h_b, BB * HD);
    for (int s = 0; s < SS; ++s) {
        gemm(h_b, encWhh_b, enc_bhh, gh, nullptr, BB, 3 * HD, HD, 0, 0, 0);
        k_gru_combine<<<(BB * HD + 255) / 256, 256, 0, stream>>>(
            gi_enc + (size_t)s * 3 * HD, SS * 3 * HD, gh, hst, h_b,
            enc_out + (size_t)s * HD, enc_out_b + (size_t)s * HD, SS * HD);
    }

    // 4) attention keys (loop-invariant): keys = enc_out @ attn_W^T + attn_b
    gemm(enc_out_b, attnW_b, attn_b, keys, nullptr, BB * SS, HD, HD, 0, 0, 0);

    // 5) decoder recurrence; defer the huge vocab projection by stashing [h2|ctx]
    for (int t = 0; t < TT - 1; ++t) {
        k_attn<<<BB, 128, 0, stream>>>(hst, keys, enc_out, ctx);
        k_build_x<<<(BB * 2048 + 255) / 256, 256, 0, stream>>>(
            tokens, t, emb, ctx, x_b, hc_b + (size_t)t * BB * 2048);
        gemm(x_b, decWih_b, dec_bih, gi_d, nullptr, BB, 3 * HD, 2 * HD, 0, 0, 0);
        gemm(h_b, decWhh_b, dec_bhh, gh_d, nullptr, BB, 3 * HD, HD, 0, 0, 0);
        k_gru_combine<<<(BB * HD + 255) / 256, 256, 0, stream>>>(
            gi_d, 3 * HD, gh_d, hst, h_b,
            nullptr, hc_b + (size_t)t * BB * 2048, 2048);
    }

    // 6) outputs[:,0,:] one-hot, then one fused vocab GEMM scattering into outputs[:,1:,:]
    k_init_out<<<(int)(((long)BB * VV + 255) / 256), 256, 0, stream>>>(out, tokens);
    gemm(hc_b, outW_b, out_b, out, nullptr, (TT - 1) * BB, VV, 2 * HD, 0, BB, TT);
}
